// GaiaProcessor_69690139345482
// MI455X (gfx1250) — matrix operand order repaired and checked by reading, not compiled, not run on hardware
//
#include <hip/hip_runtime.h>
#include <hip/hip_bf16.h>
#include <math.h>

// Problem constants (from reference)
#define NNODES 10242
#define HDIM   256
#define BATCH  2
#define KNN    8
#define NFREQ  5122      // N/2+1
#define N1F    18        // 10242 = 18 * 569
#define N2F    569

typedef __attribute__((ext_vector_type(16))) __bf16 v16bf;
typedef __attribute__((ext_vector_type(8)))  float  v8f;
typedef __attribute__((ext_vector_type(8)))  __bf16 bf16x8;

__device__ __forceinline__ __bf16 f2bf(float f) {
  union { float f; unsigned u; } v; v.f = f;
  unsigned r = v.u + 0x7fffu + ((v.u >> 16) & 1u);   // round-to-nearest-even
  unsigned short h = (unsigned short)(r >> 16);
  return __builtin_bit_cast(__bf16, h);
}
__device__ __forceinline__ float siluf(float x) { return x / (1.f + __expf(-x)); }

// ---------------------------------------------------------------------------
// Async-stage exactly 10242 contiguous floats (8-byte aligned: 10242*4 = 40968
// is 0 mod 8 for every row) from global into LDS via the CDNA5 async engine.
// 5121 x b64 transfers, ASYNCcnt-tracked; each wave drains its own count, the
// following __syncthreads() makes all waves' data visible.
// ---------------------------------------------------------------------------
__device__ __forceinline__ void gaia_async_row10242(const float* __restrict__ g,
                                                    float* lds)
{
  unsigned base = (unsigned)(uintptr_t)lds;   // low 32 bits of generic = LDS offset
  for (int i = threadIdx.x; i < 5121; i += 256) {
    unsigned la = base + ((unsigned)i << 3);
    const float* ga = g + (i << 1);
    asm volatile("global_load_async_to_lds_b64 %0, %1, off"
                 :: "v"(la), "v"(ga) : "memory");
  }
  asm volatile("s_wait_asynccnt 0x0" ::: "memory");
}

// ---------------------------------------------------------------------------
// bf16 WMMA GEMM:  C[M x 256] = act( A_concat[M x Ktot] @ W + bias )
// A0 covers k in [0,256), A1 (optional) covers k in [256,512).
// Wt: bf16, transposed weights: column n's k-run at Wt + n*ldw + k0  (k contiguous)
// Block = 128 threads = 4 waves; tile 64(M) x 64(N); grid.y = 4 (N=256 fixed).
// ---------------------------------------------------------------------------
__global__ __launch_bounds__(128)
void gaia_gemm_wmma(const float* __restrict__ A0, const float* __restrict__ A1,
                    const __bf16* __restrict__ Wt, int ldw, int k0, int Ktot,
                    const float* __restrict__ bias, float* __restrict__ C,
                    int M, int act)
{
  const int lane = threadIdx.x & 31;
  const int wave = threadIdx.x >> 5;
  const int l15  = lane & 15;
  const int half = lane >> 4;
  const int mBase = blockIdx.x * 64 + wave * 16;
  const int nBase = blockIdx.y * 64;

  v8f acc[4];
#pragma unroll
  for (int nt = 0; nt < 4; ++nt)
#pragma unroll
    for (int q = 0; q < 8; ++q) acc[nt][q] = 0.f;

  const int rowA = mBase + l15;
  const bool rowOK = rowA < M;
  const int kgA = half * 8;   // A fragment K sub-group base within 16-run

  for (int kk = 0; kk < Ktot; kk += 32) {
    // ---- A fragment (convert fp32 -> bf16 in-register) ----
    v16bf a;
#pragma unroll
    for (int g = 0; g < 2; ++g) {
      int kbase = kk + g * 16 + kgA;       // uniform halves never straddle k=256
      const float* srcp = (kbase < 256)
          ? (A0 + (size_t)rowA * 256 + kbase)
          : (A1 + (size_t)rowA * 256 + (kbase - 256));
      float4 f0 = rowOK ? *(const float4*)(srcp)     : make_float4(0.f,0.f,0.f,0.f);
      float4 f1 = rowOK ? *(const float4*)(srcp + 4) : make_float4(0.f,0.f,0.f,0.f);
      a[g*8+0] = f2bf(f0.x); a[g*8+1] = f2bf(f0.y); a[g*8+2] = f2bf(f0.z); a[g*8+3] = f2bf(f0.w);
      a[g*8+4] = f2bf(f1.x); a[g*8+5] = f2bf(f1.y); a[g*8+6] = f2bf(f1.z); a[g*8+7] = f2bf(f1.w);
      if (rowOK && kbase + 32 < Ktot) __builtin_prefetch(srcp + 32, 0, 1);
    }
    // ---- B fragments (weights already bf16, k-contiguous) ----
    v16bf bf[4];
#pragma unroll
    for (int nt = 0; nt < 4; ++nt) {
      int col = nBase + nt * 16 + l15;
      const __bf16* wp = Wt + (size_t)col * ldw + k0 + kk + half * 8;
      bf16x8 w0 = *(const bf16x8*)(wp);
      bf16x8 w1 = *(const bf16x8*)(wp + 16);
#pragma unroll
      for (int e = 0; e < 8; ++e) { bf[nt][e] = w0[e]; bf[nt][8+e] = w1[e]; }
    }
#pragma unroll
    for (int nt = 0; nt < 4; ++nt)
      acc[nt] = __builtin_amdgcn_wmma_f32_16x16x32_bf16(
          false, a, false, bf[nt], (short)0, acc[nt], false, false);
  }

#pragma unroll
  for (int nt = 0; nt < 4; ++nt) {
    int col = nBase + nt * 16 + l15;
    float bv = bias ? bias[col] : 0.f;
#pragma unroll
    for (int v = 0; v < 8; ++v) {
      int row = mBase + half * 8 + v;
      if (row < M) {
        float r = acc[nt][v] + bv;
        if (act) r = siluf(r);
        C[(size_t)row * 256 + col] = r;
      }
    }
  }
}

// ---------------------------------------------------------------------------
// Convert W[K x 256] fp32 (row-major) -> Wt[256 x K] bf16 (k-contiguous)
// ---------------------------------------------------------------------------
__global__ void gaia_wprep(const float* __restrict__ W, __bf16* __restrict__ Wt, int K)
{
  int idx = blockIdx.x * 256 + threadIdx.x;
  if (idx < K * 256) {
    int n = idx / K, k = idx - n * K;
    Wt[idx] = f2bf(W[(size_t)k * 256 + n]);
  }
}

// lat hidden: silu(lat[n]*w1[h] + b1[h])  (din == 1)
__global__ void gaia_lat_hidden(const float* __restrict__ lat, const float* __restrict__ w1,
                                const float* __restrict__ b1, float* __restrict__ out)
{
  int idx = blockIdx.x * 256 + threadIdx.x;
  if (idx < NNODES * HDIM) {
    int n = idx >> 8, h = idx & 255;
    out[idx] = siluf(lat[n] * w1[h] + b1[h]);
  }
}

// hbar[b,n,h] = mean_k silu(P[b,n,h] + Q[b,nbr,h] + b1[h])   (mean commutes with w2)
__global__ void gaia_msg_combine(const float* __restrict__ P, const float* __restrict__ Q,
                                 const float* __restrict__ b1, const int* __restrict__ knn,
                                 float* __restrict__ hbar, int Nn)
{
  int row = blockIdx.x;                 // b*Nn + n
  int h = threadIdx.x;
  int n = row % Nn, b = row / Nn;
  float pv = P[(size_t)row * HDIM + h] + b1[h];
  float acc = 0.f;
#pragma unroll
  for (int k = 0; k < KNN; ++k) {
    int nbr = knn[n * KNN + k];
    float t = pv + Q[((size_t)b * Nn + nbr) * HDIM + h];
    acc += t / (1.f + __expf(-t));
  }
  hbar[(size_t)row * HDIM + h] = acc * (1.f / KNN);
}

// Fused residual + lat_bias (+ per-batch glob vec) + LayerNorm. 1 wave per row.
__global__ __launch_bounds__(256)
void gaia_ln(const float* __restrict__ x, const float* __restrict__ add1,
             const float* __restrict__ latb, const float* __restrict__ gvec,
             const float* __restrict__ gam, const float* __restrict__ bet,
             float* __restrict__ out, int rows, int Nn)
{
  int row = blockIdx.x * 8 + (threadIdx.x >> 5);
  int lane = threadIdx.x & 31;
  if (row >= rows) return;
  int b = row / Nn, n = row - b * Nn;
  const size_t ro = (size_t)row * HDIM;
  float v[8]; float s = 0.f;
#pragma unroll
  for (int j = 0; j < 8; ++j) {
    int h = lane + 32 * j;
    float t = x[ro + h] + add1[ro + h] + latb[(size_t)n * HDIM + h];
    if (gvec) t += gvec[b * HDIM + h];
    v[j] = t; s += t;
  }
#pragma unroll
  for (int off = 16; off > 0; off >>= 1) s += __shfl_xor(s, off, 32);
  float m = s * (1.f / HDIM);
  float q = 0.f;
#pragma unroll
  for (int j = 0; j < 8; ++j) { float d = v[j] - m; q += d * d; }
#pragma unroll
  for (int off = 16; off > 0; off >>= 1) q += __shfl_xor(q, off, 32);
  float inv = rsqrtf(q * (1.f / HDIM) + 1e-5f);
#pragma unroll
  for (int j = 0; j < 8; ++j) {
    int h = lane + 32 * j;
    out[ro + h] = (v[j] - m) * inv * gam[h] + bet[h];
  }
}

// ---------------------------------------------------------------------------
// LDS-tiled transpose with optional node permutation.
//  mode 0: out[p][c][r] = in[p][r][c]
//  mode 1: out[p][c][r] = in[p][perm[r]][c]     (gather rows)
//  mode 2: out[p][perm[r]][c] = in[p][c][r]     (scatter rows; in is (P,C,R))
// ---------------------------------------------------------------------------
__global__ __launch_bounds__(256)
void gaia_transpose(const float* __restrict__ in, float* __restrict__ out,
                    int R, int C, const int* __restrict__ perm, int mode)
{
  __shared__ float tile[32][33];
  int p = blockIdx.z;
  int r0 = blockIdx.x * 32, c0 = blockIdx.y * 32;
  int tx = threadIdx.x & 31, ty = threadIdx.x >> 5;
  if (mode <= 1) {
    for (int i = ty; i < 32; i += 8) {
      int r = r0 + i, c = c0 + tx;
      if (r < R && c < C) {
        int rs = (mode == 1) ? perm[r] : r;
        tile[i][tx] = in[((size_t)p * R + rs) * C + c];
      }
    }
    __syncthreads();
    for (int i = ty; i < 32; i += 8) {
      int c = c0 + i, r = r0 + tx;
      if (r < R && c < C) out[((size_t)p * C + c) * R + r] = tile[tx][i];
    }
  } else {
    for (int i = ty; i < 32; i += 8) {
      int c = c0 + i, r = r0 + tx;
      if (r < R && c < C) tile[i][tx] = in[((size_t)p * C + c) * R + r];
    }
    __syncthreads();
    for (int i = ty; i < 32; i += 8) {
      int r = r0 + i, c = c0 + tx;
      if (r < R && c < C) {
        int rd = perm ? perm[r] : r;
        out[((size_t)p * R + rd) * C + c] = tile[tx][i];
      }
    }
  }
}

// Forward twiddle table W_N[j] = exp(-2*pi*i*j/N)
__global__ void gaia_build_wn(float2* __restrict__ WN)
{
  int j = blockIdx.x * 256 + threadIdx.x;
  if (j < NNODES) {
    float s, c;
    __sincosf(-6.283185307179586f * (float)j / (float)NNODES, &s, &c);
    WN[j] = make_float2(c, s);
  }
}

// FFT stage 1 (forward): 18 DFT-569 per (b,h) row; row async-staged into LDS
// (41KB) and waited on, then twiddles are computed, then LDS-fed inner DFT loops.
__global__ __launch_bounds__(256)
void gaia_fft_f1(const float* __restrict__ xsT, float2* __restrict__ Y)
{
  int bh = blockIdx.x;
  __shared__ float xrow[NNODES];
  __shared__ float2 w569[N2F];
  const float* xr = xsT + (size_t)bh * NNODES;
  gaia_async_row10242(xr, xrow);           // ASYNCcnt path: global -> LDS direct
  for (int i = threadIdx.x; i < N2F; i += 256) {
    float s, c;
    __sincosf(-6.283185307179586f * (float)i / (float)N2F, &s, &c);
    w569[i] = make_float2(c, s);
  }
  __syncthreads();
  for (int o = threadIdx.x; o < N1F * N2F; o += 256) {
    int n1 = o / N2F, k2 = o - n1 * N2F;
    float ar = 0.f, ai = 0.f; int j = 0;
    for (int n2 = 0; n2 < N2F; ++n2) {
      float xv = xrow[n1 + N1F * n2];
      float2 w = w569[j];
      ar += xv * w.x; ai += xv * w.y;
      j += k2; if (j >= N2F) j -= N2F;
    }
    Y[((size_t)bh * N1F + n1) * N2F + k2] = make_float2(ar, ai);
  }
}

// FFT stage 2 (forward): twiddle + 18-point combine, keep rfft bins [0,5122).
__global__ __launch_bounds__(256)
void gaia_fft_f2(const float2* __restrict__ Y, const float2* __restrict__ WN,
                 float* __restrict__ FrT, float* __restrict__ FiT)
{
  int bh = blockIdx.x;
  __shared__ float2 w18[N1F];
  if (threadIdx.x < N1F) {
    float s, c;
    __sincosf(-6.283185307179586f * (float)threadIdx.x / (float)N1F, &s, &c);
    w18[threadIdx.x] = make_float2(c, s);
  }
  __syncthreads();
  const float2* y = Y + (size_t)bh * N1F * N2F;
  for (int k = threadIdx.x; k < NFREQ; k += 256) {
    int k1 = k / N2F, k2 = k - k1 * N2F;
    float ar = 0.f, ai = 0.f; int j = 0;   // j = n1*k2 mod N
    for (int n1 = 0; n1 < N1F; ++n1) {
      float2 yv = y[n1 * N2F + k2];
      float2 wa = WN[j];
      float2 wb = w18[(n1 * k1) % N1F];
      float wr = wa.x * wb.x - wa.y * wb.y;
      float wi = wa.x * wb.y + wa.y * wb.x;
      ar += yv.x * wr - yv.y * wi;
      ai += yv.x * wi + yv.y * wr;
      j += k2; if (j >= NNODES) j -= NNODES;
    }
    FrT[(size_t)bh * NFREQ + k] = ar;
    FiT[(size_t)bh * NFREQ + k] = ai;
  }
}

// Inverse stage 1: Hermitian-extend spectrum, 18-point DFTs (positive sign).
__global__ __launch_bounds__(256)
void gaia_fft_i1(const float* __restrict__ Rr, const float* __restrict__ Ri,
                 float2* __restrict__ Z)
{
  int bh = blockIdx.x;
  __shared__ float2 w18[N1F];
  if (threadIdx.x < N1F) {
    float s, c;
    __sincosf(6.283185307179586f * (float)threadIdx.x / (float)N1F, &s, &c);
    w18[threadIdx.x] = make_float2(c, s);
  }
  __syncthreads();
  const float* rr = Rr + (size_t)bh * NFREQ;
  const float* ri = Ri + (size_t)bh * NFREQ;
  for (int o = threadIdx.x; o < N1F * N2F; o += 256) {
    int m = o / N2F, k2 = o - m * N2F;
    float ar = 0.f, ai = 0.f;
    for (int k1 = 0; k1 < N1F; ++k1) {
      int k = k2 + N2F * k1;
      float sr, si;
      if (k < NFREQ) { sr = rr[k]; si = ri[k]; }
      else { int kc = NNODES - k; sr = rr[kc]; si = -ri[kc]; }
      float2 w = w18[(m * k1) % N1F];
      ar += sr * w.x - si * w.y;
      ai += sr * w.y + si * w.x;
    }
    Z[((size_t)bh * N1F + m) * N2F + k2] = make_float2(ar, ai);
  }
}

// Inverse stage 2: 569-point inverse combine, real output, 9 m-rows per block.
// Z slab (9*569 complex = exactly 10242 floats, 8B aligned) async-staged to LDS.
__global__ __launch_bounds__(256)
void gaia_fft_i2(const float2* __restrict__ Z, const float2* __restrict__ WN,
                 float* __restrict__ y)
{
  int bh = blockIdx.x;
  int m0 = blockIdx.y * 9;
  __shared__ float2 zsh[9 * N2F];
  const float2* z = Z + ((size_t)bh * N1F + m0) * N2F;
  gaia_async_row10242((const float*)z, (float*)zsh);   // ASYNCcnt path
  __syncthreads();
  const float invN = 1.f / (float)NNODES;
  for (int o = threadIdx.x; o < 9 * N2F; o += 256) {
    int mm = o / N2F, q = o - mm * N2F;
    int s = (m0 + mm) + N1F * q;
    float ar = 0.f; int j = 0;            // j = s*k2 mod N; inverse = conj(WN)
    for (int k2 = 0; k2 < N2F; ++k2) {
      float2 zv = zsh[mm * N2F + k2];
      float2 w = WN[j];
      ar += zv.x * w.x + zv.y * w.y;      // Re(z * e^{+i theta})
      j += s; if (j >= NNODES) j -= NNODES;
    }
    y[(size_t)bh * NNODES + s] = ar * invN;
  }
}

// Deterministic mean over nodes: partials then reduce (no float atomics).
__global__ void gaia_mean_part(const float* __restrict__ x, float* __restrict__ part, int Nn)
{
  int b = blockIdx.x, chunk = blockIdx.y, h = threadIdx.x;
  int per = (Nn + 63) / 64;
  int n0 = chunk * per, n1 = n0 + per; if (n1 > Nn) n1 = Nn;
  float s = 0.f;
  for (int n = n0; n < n1; ++n) s += x[((size_t)b * Nn + n) * HDIM + h];
  part[((size_t)b * 64 + chunk) * HDIM + h] = s;
}
__global__ void gaia_mean_red(const float* __restrict__ part, float* __restrict__ xsum)
{
  int b = blockIdx.x, h = threadIdx.x;
  float s = 0.f;
  for (int c = 0; c < 64; ++c) s += part[((size_t)b * 64 + c) * HDIM + h];
  xsum[b * HDIM + h] = s;
}

// Tiny glob MLP (2 rows): keep fp32 VALU.
__global__ void gaia_glob(const float* __restrict__ xsum,
                          const float* __restrict__ w1, const float* __restrict__ b1,
                          const float* __restrict__ w2, const float* __restrict__ b2,
                          float* __restrict__ g, float invN)
{
  int b = blockIdx.x, h = threadIdx.x;
  __shared__ float xin[HDIM], hid[HDIM];
  xin[h] = xsum[b * HDIM + h] * invN;
  __syncthreads();
  float a = b1[h];
  for (int i = 0; i < HDIM; ++i) a += xin[i] * w1[i * HDIM + h];
  hid[h] = siluf(a);
  __syncthreads();
  float o = b2[h];
  for (int i = 0; i < HDIM; ++i) o += hid[i] * w2[i * HDIM + h];
  g[b * HDIM + h] = o;
}

// ---------------------------------------------------------------------------
extern "C" void kernel_launch(void* const* d_in, const int* in_sizes, int n_in,
                              void* d_out, int out_size, void* d_ws, size_t ws_size,
                              hipStream_t stream)
{
  (void)n_in; (void)out_size;
  const int Nn = NNODES, Hh = HDIM, Bb = BATCH;
  const int BN = Bb * Nn;                 // 20484 rows
  const int BNf = Bb * NFREQ;             // 10244 rows
  const size_t NH   = (size_t)Nn * Hh;
  const size_t BNH  = (size_t)Bb * NH;
  const size_t BNfH = (size_t)Bb * NFREQ * Hh;

  auto Fp = [&](int i) { return (const float*)d_in[i]; };
  auto Ip = [&](int i) { return (const int*)d_in[i]; };

  struct MlpP { const float *w1, *b1, *w2, *b2; };
  struct BlkP { MlpP m1, m2; const float *g, *b; };   // gnn: msg/upd ; spec: freq/glob
  const float *x0, *lat; const int *knn, *order;
  MlpP lb; BlkP blk[8];

  if (in_sizes[0] == (int)BNH) {
    // insertion order: x, lat, lat_bias{w1,b1,w2,b2}, blocks[8]{...}, knn, order
    x0 = Fp(0); lat = Fp(1);
    lb = { Fp(2), Fp(3), Fp(4), Fp(5) };
    int i = 6;
    for (int bI = 0; bI < 8; ++bI) {
      blk[bI].m1 = { Fp(i), Fp(i+1), Fp(i+2), Fp(i+3) }; i += 4;
      blk[bI].m2 = { Fp(i), Fp(i+1), Fp(i+2), Fp(i+3) }; i += 4;
      blk[bI].g = Fp(i); blk[bI].b = Fp(i+1); i += 2;
    }
    knn = Ip(86); order = Ip(87);
  } else {
    // sorted pytree order: knn, lat, order, blocks[8], lat_bias{b1,b2,w1,w2}, x
    knn = Ip(0); lat = Fp(1); order = Ip(2);
    int i = 3;
    for (int bI = 0; bI < 8; ++bI) {
      if ((bI & 1) == 0) { // gnn: ln.b, ln.g, msg.b1,b2,w1,w2, upd.b1,b2,w1,w2
        blk[bI].b = Fp(i); blk[bI].g = Fp(i+1);
        blk[bI].m1 = { Fp(i+4), Fp(i+2), Fp(i+5), Fp(i+3) };
        blk[bI].m2 = { Fp(i+8), Fp(i+6), Fp(i+9), Fp(i+7) };
      } else {            // spec: freq.b1,b2,w1,w2, glob.b1,b2,w1,w2, ln.b, ln.g
        blk[bI].m1 = { Fp(i+2), Fp(i),   Fp(i+3), Fp(i+1) };
        blk[bI].m2 = { Fp(i+6), Fp(i+4), Fp(i+7), Fp(i+5) };
        blk[bI].b = Fp(i+8); blk[bI].g = Fp(i+9);
      }
      i += 10;
    }
    lb = { Fp(i+2), Fp(i), Fp(i+3), Fp(i+1) }; i += 4;
    x0 = Fp(i);
  }

  // ---- workspace layout (fp32 bump allocator) ----
  float* W = (float*)d_ws;
  size_t off = 0;
  auto alloc = [&](size_t n) { size_t o = off; off += (n + 63) & ~(size_t)63; return o; };
  size_t o_latT = alloc(NH), o_latb = alloc(NH);
  size_t o_xA = alloc(BNH), o_xB = alloc(BNH);
  size_t o_t0 = alloc(BNH), o_t1 = alloc(BNH), o_t2 = alloc(BNH);
  size_t o_Y  = alloc(2 * BNH);                         // complex (B,H,N)
  size_t o_Fr = alloc(BNfH), o_Fi = alloc(BNfH);
  size_t o_FMr = alloc(BNfH), o_FMi = alloc(BNfH), o_Fh = alloc(BNfH);
  size_t o_WN = alloc(2 * Nn);
  size_t o_part = alloc((size_t)Bb * 64 * Hh);
  size_t o_xsum = alloc((size_t)Bb * Hh);
  size_t o_gv = alloc((size_t)Bb * Hh);
  const size_t bfElems = 65536 + 4 * (131072 + 65536 + 131072 + 65536) + 4 * (65536 + 65536);
  size_t o_wbf = alloc(bfElems / 2 + 64);
  if (ws_size < off * sizeof(float)) return;            // scratch too small: bail

  __bf16* wbase = (__bf16*)(W + o_wbf);
  size_t wo = 0;
  auto walloc = [&](size_t n) { __bf16* p = wbase + wo; wo += n; return p; };
  __bf16 *latW2t = walloc(65536);
  __bf16 *w1t[8], *w2t[8], *w3t[8], *w4t[8];
  for (int bI = 0; bI < 8; ++bI) {
    if ((bI & 1) == 0) {
      w1t[bI] = walloc(131072); w2t[bI] = walloc(65536);
      w3t[bI] = walloc(131072); w4t[bI] = walloc(65536);
    } else {
      w1t[bI] = walloc(65536);  w2t[bI] = walloc(65536);
      w3t[bI] = nullptr; w4t[bI] = nullptr;
    }
  }

  auto prep = [&](const float* Wsrc, __bf16* Wt, int K) {
    int total = K * 256;
    gaia_wprep<<<dim3((total + 255) / 256), dim3(256), 0, stream>>>(Wsrc, Wt, K);
  };
  auto gemm = [&](const float* A0, const float* A1, const __bf16* Wt, int ldw, int k0,
                  int Ktot, const float* bias, float* Cm, int M, int act) {
    gaia_gemm_wmma<<<dim3((M + 63) / 64, 4), dim3(128), 0, stream>>>(
        A0, A1, Wt, ldw, k0, Ktot, bias, Cm, M, act);
  };
  auto tgrid = [](int R, int C, int P) { return dim3((R + 31) / 32, (C + 31) / 32, P); };

  // ---- one-time per launch: twiddles + bf16 weight prep ----
  gaia_build_wn<<<dim3((Nn + 255) / 256), dim3(256), 0, stream>>>((float2*)(W + o_WN));
  prep(lb.w2, latW2t, 256);
  for (int bI = 0; bI < 8; ++bI) {
    if ((bI & 1) == 0) {
      prep(blk[bI].m1.w1, w1t[bI], 512); prep(blk[bI].m1.w2, w2t[bI], 256);
      prep(blk[bI].m2.w1, w3t[bI], 512); prep(blk[bI].m2.w2, w4t[bI], 256);
    } else {
      prep(blk[bI].m1.w1, w1t[bI], 256); prep(blk[bI].m1.w2, w2t[bI], 256);
    }
  }

  // ---- lat_bias MLP (shared across batch) ----
  gaia_lat_hidden<<<dim3((int)((NH + 255) / 256)), dim3(256), 0, stream>>>(
      lat, lb.w1, lb.b1, W + o_latT);
  gemm(W + o_latT, nullptr, latW2t, 256, 0, 256, lb.b2, W + o_latb, Nn, 0);

  // ---- 8 blocks, ping-pong x ----
  const float* xin = x0;
  for (int bI = 0; bI < 8; ++bI) {
    float* dst = (bI == 7) ? (float*)d_out : (W + (((bI & 1) == 0) ? o_xA : o_xB));
    if ((bI & 1) == 0) {
      // ======= GNN block =======
      // P = X @ W1[:256,:],  Q = X @ W1[256:,:]
      gemm(xin, nullptr, w1t[bI], 512, 0,   256, nullptr, W + o_t0, BN, 0);
      gemm(xin, nullptr, w1t[bI], 512, 256, 256, nullptr, W + o_t1, BN, 0);
      gaia_msg_combine<<<dim3(BN), dim3(256), 0, stream>>>(
          W + o_t0, W + o_t1, blk[bI].m1.b1, knn, W + o_t2, Nn);
      // msg = mean_k(silu) @ W2 + b2   (mean commutes with linear layer)
      gemm(W + o_t2, nullptr, w2t[bI], 256, 0, 256, blk[bI].m1.b2, W + o_t0, BN, 0);
      // upd = silu(X@W1a + MSG@W1b + b1) @ W2 + b2
      gemm(xin, W + o_t0, w3t[bI], 512, 0, 512, blk[bI].m2.b1, W + o_t1, BN, 1);
      gemm(W + o_t1, nullptr, w4t[bI], 256, 0, 256, blk[bI].m2.b2, W + o_t2, BN, 0);
      gaia_ln<<<dim3((BN + 7) / 8), dim3(256), 0, stream>>>(
          xin, W + o_t2, W + o_latb, nullptr, blk[bI].g, blk[bI].b, dst, BN, Nn);
    } else {
      // ======= Spectral block =======
      gaia_transpose<<<tgrid(Nn, Hh, Bb), dim3(256), 0, stream>>>(
          xin, W + o_t0, Nn, Hh, order, 1);                       // gather+transpose -> (B,H,N)
      gaia_fft_f1<<<dim3(Bb * Hh), dim3(256), 0, stream>>>(W + o_t0, (float2*)(W + o_Y));
      gaia_fft_f2<<<dim3(Bb * Hh), dim3(256), 0, stream>>>(
          (const float2*)(W + o_Y), (const float2*)(W + o_WN), W + o_Fr, W + o_Fi);
      gaia_transpose<<<tgrid(Hh, NFREQ, Bb), dim3(256), 0, stream>>>(
          W + o_Fr, W + o_FMr, Hh, NFREQ, nullptr, 0);            // (B,H,k)->(B,k,H)
      gaia_transpose<<<tgrid(Hh, NFREQ, Bb), dim3(256), 0, stream>>>(
          W + o_Fi, W + o_FMi, Hh, NFREQ, nullptr, 0);
      // freq MLP (same weights for real and imag)
      gemm(W + o_FMr, nullptr, w1t[bI], 256, 0, 256, blk[bI].m1.b1, W + o_Fh, BNf, 1);
      gemm(W + o_Fh,  nullptr, w2t[bI], 256, 0, 256, blk[bI].m1.b2, W + o_Fr, BNf, 0);
      gemm(W + o_FMi, nullptr, w1t[bI], 256, 0, 256, blk[bI].m1.b1, W + o_Fh, BNf, 1);
      gemm(W + o_Fh,  nullptr, w2t[bI], 256, 0, 256, blk[bI].m1.b2, W + o_Fi, BNf, 0);
      gaia_transpose<<<tgrid(NFREQ, Hh, Bb), dim3(256), 0, stream>>>(
          W + o_Fr, W + o_FMr, NFREQ, Hh, nullptr, 0);            // (B,k,H)->(B,H,k)
      gaia_transpose<<<tgrid(NFREQ, Hh, Bb), dim3(256), 0, stream>>>(
          W + o_Fi, W + o_FMi, NFREQ, Hh, nullptr, 0);
      gaia_fft_i1<<<dim3(Bb * Hh), dim3(256), 0, stream>>>(
          W + o_FMr, W + o_FMi, (float2*)(W + o_Y));
      gaia_fft_i2<<<dim3(Bb * Hh, 2), dim3(256), 0, stream>>>(
          (const float2*)(W + o_Y), (const float2*)(W + o_WN), W + o_t1);
      gaia_transpose<<<tgrid(Nn, Hh, Bb), dim3(256), 0, stream>>>(
          W + o_t1, W + o_t2, Nn, Hh, order, 2);                  // scatter back -> (B,N,H)
      // glob MLP on node-mean (deterministic two-stage reduce)
      gaia_mean_part<<<dim3(Bb, 64), dim3(256), 0, stream>>>(xin, W + o_part, Nn);
      gaia_mean_red<<<dim3(Bb), dim3(256), 0, stream>>>(W + o_part, W + o_xsum);
      gaia_glob<<<dim3(Bb), dim3(256), 0, stream>>>(
          W + o_xsum, blk[bI].m2.w1, blk[bI].m2.b1, blk[bI].m2.w2, blk[bI].m2.b2,
          W + o_gv, 1.f / (float)Nn);
      gaia_ln<<<dim3((BN + 7) / 8), dim3(256), 0, stream>>>(
          xin, W + o_t2, W + o_latb, W + o_gv, blk[bI].g, blk[bI].b, dst, BN, Nn);
    }
    xin = dst;
  }
}
